// XLNetRelativeAttention_65214783422619
// MI455X (gfx1250) — matrix operand order repaired and checked by reading, not compiled, not run on hardware
//
#include <hip/hip_runtime.h>

// ---------------------------------------------------------------------------
// XLNet relative attention for MI455X (gfx1250, wave32, WMMA bf16, async LDS).
//   1) cvt f32->bf16 : h, g, r, Wq, Wk, Wv, Wr ; Wo transposed during cvt
//   2) pack_code     : (mask_h | mask_g<<1 | seg<<2) -> u8[b][i][j]
//   3) gemm_bf16     : q_h, q_g (f32 out), k_h, v_h, k_r (bf16 out)
//                      double-buffered global_load_async_to_lds_b128 staging
//   4) attn_core     : fused ac + banded-bd (rel_shift) + ef + softmax + PV
//   5) gemm_bf16     : output projection with pre-transposed Wo
//   6) residual_ln   : +inp, LayerNorm -> d_out
// ---------------------------------------------------------------------------

#define QLEN 512
#define BATCH 8
#define DMODEL 1024
#define NHEAD 16
#define KLEN 512
#define MROWS (QLEN * BATCH)     // 4096
#define MROWS_R (1024 * BATCH)   // 8192
#define ATT_SCALE 0.125f         // 1/sqrt(64)

typedef __attribute__((ext_vector_type(16))) __bf16 v16bf;
typedef __attribute__((ext_vector_type(8)))  float  v8f;

union FragU {
    v16bf v;
    unsigned short s[16];
};

__device__ __forceinline__ unsigned short f2bf(float x) {
    unsigned int u = __float_as_uint(x);
    unsigned int r = u + 0x7FFFu + ((u >> 16) & 1u);   // round-to-nearest-even
    return (unsigned short)(r >> 16);
}

__device__ __forceinline__ v8f wmma_bf16(v16bf a, v16bf b, v8f c) {
    return __builtin_amdgcn_wmma_f32_16x16x32_bf16(
        false, a, false, b, (short)0, c, false, false);
}

// gfx1250 async global->LDS copy: 16 bytes per lane, tracked by ASYNCcnt.
__device__ __forceinline__ void async_load_b128(unsigned lds_off, const void* gptr) {
    asm volatile("global_load_async_to_lds_b128 %0, %1, off"
                 :: "v"(lds_off), "v"(gptr) : "memory");
}
__device__ __forceinline__ void wait_async0() {
    asm volatile("s_wait_asynccnt 0x0" ::: "memory");
}
__device__ __forceinline__ void wait_async4() {   // prev stage (4 issues) done
    asm volatile("s_wait_asynccnt 0x4" ::: "memory");
}
__device__ __forceinline__ void wait_async1() {   // prev stage (1 issue) done
    asm volatile("s_wait_asynccnt 0x1" ::: "memory");
}
__device__ __forceinline__ unsigned lds_addr_of(const void* p) {
    return (unsigned)(unsigned long long)p;   // low 32 bits = LDS offset
}

// A fragment: 16x32 bf16, M = lane%16; per-lane K covers {0..7,16..23} (+8 hi)
// -> two contiguous 16B runs per lane (compiler emits 2x ds_load_b128).
__device__ __forceinline__ v16bf load_frag_a(const unsigned short* base,
                                             int stride, int lane) {
    FragU f;
    int m  = lane & 15;
    int hi = (lane >> 4) & 1;
#pragma unroll
    for (int e = 0; e < 16; ++e) {
        int v = e >> 1;
        int k = ((v >> 2) << 4) + (hi << 3) + ((v & 3) << 1) + (e & 1);
        f.s[e] = base[m * stride + k];
    }
    return f.v;
}

// B fragment: 32x16 bf16, N = lane%16; per-lane K covers {0..7,16..23} (+8 hi).
__device__ __forceinline__ v16bf load_frag_b(const unsigned short* base,
                                             int stride, int lane) {
    FragU f;
    int nn = lane & 15;
    int hi = (lane >> 4) & 1;
#pragma unroll
    for (int e = 0; e < 16; ++e) {
        int k = ((e >> 3) << 4) + (hi << 3) + (e & 7);
        f.s[e] = base[k * stride + nn];
    }
    return f.v;
}

// ---------------------------------------------------------------------------
__global__ __launch_bounds__(256) void cvt_f32_bf16(
    const float* __restrict__ x, unsigned short* __restrict__ y, int n) {
    int i = (blockIdx.x * 256 + threadIdx.x) * 4;
    if (i + 3 < n) {
#pragma unroll
        for (int k = 0; k < 4; ++k) y[i + k] = f2bf(x[i + k]);
    }
}

// Transpose-convert 1024x1024: y[p][h] = bf16(x[h][p]); 32x32 LDS tiles.
__global__ __launch_bounds__(256) void cvt_transpose_bf16(
    const float* __restrict__ x, unsigned short* __restrict__ y) {
    __shared__ float tile[32][33];
    int tx = (blockIdx.x & 31) * 32;   // p base (src col)
    int ty = (blockIdx.x >> 5) * 32;   // h base (src row)
    int c = threadIdx.x & 31, r0 = threadIdx.x >> 5;
#pragma unroll
    for (int k = 0; k < 4; ++k) {
        int r = r0 + k * 8;
        tile[r][c] = x[(size_t)(ty + r) * 1024 + tx + c];
    }
    __syncthreads();
#pragma unroll
    for (int k = 0; k < 4; ++k) {
        int r = r0 + k * 8;            // r -> p offset, c -> h offset
        y[(size_t)(tx + r) * 1024 + ty + c] = f2bf(tile[c][r]);
    }
}

// ---------------------------------------------------------------------------
__global__ __launch_bounds__(256) void pack_code(
    const float* __restrict__ mh, const float* __restrict__ mg,
    const float* __restrict__ segm, unsigned char* __restrict__ code) {
    int idx = blockIdx.x * 256 + threadIdx.x;      // == ((i*K + j)*B + b)
    int i = idx >> 12;
    int j = (idx >> 3) & 511;
    int b = idx & 7;
    unsigned char c = 0;
    if (mh[idx] != 0.f)           c |= 1;
    if (mg[idx] != 0.f)           c |= 2;
    if (segm[2 * idx + 1] != 0.f) c |= 4;
    code[(b * QLEN + i) * KLEN + j] = c;
}

// ---------------------------------------------------------------------------
// bf16 GEMM: C[M][1024] = A[M][1024] x B[1024][1024], both row-major bf16.
// 128x128 tile, 8 waves x (32x64). Double-buffered async LDS staging:
// prefetch step i+1 while computing step i (s_wait_asynccnt <= 4).
// ---------------------------------------------------------------------------
__global__ __launch_bounds__(256) void gemm_bf16_128x128(
    const unsigned short* __restrict__ A, const unsigned short* __restrict__ Bm,
    float* __restrict__ Cf, unsigned short* __restrict__ Cb) {
    __shared__ __attribute__((aligned(16))) unsigned short A_s[2][128 * 40];
    __shared__ __attribute__((aligned(16))) unsigned short B_s[2][32 * 136];

    const int tid  = threadIdx.x;
    const int lane = tid & 31;
    const int wave = tid >> 5;
    const int wm   = wave & 3;   // 4 groups of 32 rows
    const int wp   = wave >> 2;  // 2 groups of 64 cols
    const int p0   = blockIdx.x * 128;
    const int m0   = blockIdx.y * 128;

    // per-thread staging chunks (2 for A, 2 for B)
    const int chA  = tid * 2;
    const int rA0  = chA >> 2,        cA0 = (chA & 3) * 16;
    const int rA1  = (chA + 1) >> 2,  cA1 = ((chA + 1) & 3) * 16;
    const int kB0  = chA >> 4,        cB0 = (chA & 15) * 16;
    const int kB1  = (chA + 1) >> 4,  cB1 = ((chA + 1) & 15) * 16;

    const char* aP0 = (const char*)A + (((size_t)(m0 + rA0)) << 11) + cA0;
    const char* aP1 = (const char*)A + (((size_t)(m0 + rA1)) << 11) + cA1;
    const char* bP0 = (const char*)Bm + (((size_t)kB0) << 11) + p0 * 2 + cB0;
    const char* bP1 = (const char*)Bm + (((size_t)kB1) << 11) + p0 * 2 + cB1;

    const unsigned aLds[2] = { lds_addr_of(A_s[0]), lds_addr_of(A_s[1]) };
    const unsigned bLds[2] = { lds_addr_of(B_s[0]), lds_addr_of(B_s[1]) };
    const unsigned offA0 = rA0 * 80 + cA0, offA1 = rA1 * 80 + cA1;
    const unsigned offB0 = kB0 * 272 + cB0, offB1 = kB1 * 272 + cB1;

    v8f acc[2][4] = {};

    // prologue: stage step 0 into buffer 0
    async_load_b128(aLds[0] + offA0, aP0);
    async_load_b128(aLds[0] + offA1, aP1);
    async_load_b128(bLds[0] + offB0, bP0);
    async_load_b128(bLds[0] + offB1, bP1);
    aP0 += 64; aP1 += 64; bP0 += 65536; bP1 += 65536;

    for (int itr = 0; itr < 32; ++itr) {
        const int cur = itr & 1;
        if (itr != 31) {
            const int nxt = cur ^ 1;
            async_load_b128(aLds[nxt] + offA0, aP0);
            async_load_b128(aLds[nxt] + offA1, aP1);
            async_load_b128(bLds[nxt] + offB0, bP0);
            async_load_b128(bLds[nxt] + offB1, bP1);
            aP0 += 64; aP1 += 64; bP0 += 65536; bP1 += 65536;
            wait_async4();
        } else {
            wait_async0();
        }
        __syncthreads();

        const unsigned short* As = A_s[cur];
        const unsigned short* Bs = B_s[cur];
        v16bf af[2], bf[4];
#pragma unroll
        for (int ms = 0; ms < 2; ++ms)
            af[ms] = load_frag_a(As + (wm * 32 + ms * 16) * 40, 40, lane);
#pragma unroll
        for (int ps = 0; ps < 4; ++ps)
            bf[ps] = load_frag_b(Bs + wp * 64 + ps * 16, 136, lane);
#pragma unroll
        for (int ms = 0; ms < 2; ++ms)
#pragma unroll
            for (int ps = 0; ps < 4; ++ps)
                acc[ms][ps] = wmma_bf16(af[ms], bf[ps], acc[ms][ps]);
        __syncthreads();
    }

    const int hi8 = ((lane >> 4) & 1) << 3;
#pragma unroll
    for (int ms = 0; ms < 2; ++ms)
#pragma unroll
        for (int ps = 0; ps < 4; ++ps)
#pragma unroll
            for (int r = 0; r < 8; ++r) {
                int m = m0 + wm * 32 + ms * 16 + r + hi8;
                int p = p0 + wp * 64 + ps * 16 + (lane & 15);
                float v = acc[ms][ps][r];
                if (Cb) Cb[((size_t)m << 10) + p] = f2bf(v);
                else    Cf[((size_t)m << 10) + p] = v;
            }
}

// ---------------------------------------------------------------------------
// Fused attention core: one workgroup per (b, n, 64-row q tile).
// Scores computed TRANSPOSED (K / Kr band are the A operand and V the B operand, straight from
// global layout; qw/qr built transposed in LDS), so every global->LDS stage
// is an async b128 row copy.
// ---------------------------------------------------------------------------
// dynamic LDS layout (bytes)
#define OFF_QF32  0         // 64 x 68 f32   = 17408
#define OFF_QWT   17408     // 64 x 72 u16   = 9216   (q + r_w_bias)^T [d][i]
#define OFF_QRT   26624     // 64 x 72 u16   = 9216   (q + r_r_bias)^T [d][i]
#define OFF_EF    35840     // 128 f32       = 512
#define OFF_KS    36352     // 64 x 72 u16   = 9216   K rows [j][d] (V ping)
#define OFF_KRS   45568     // 128 x 72 u16  = 18432  Kr band [pp][d] (V pong)
#define OFF_BDF   64000     // 128 x 68 f32  = 34816  bd^T band [pp][i]
#define OFF_SC    98816     // 64 x 516 f32  = 132096
#define OFF_PB    230912    // 64 x 520 u16  = 66560
#define OFF_RA    297472    // 256 f32
#define OFF_RB    298496    // 256 f32
#define ATT_SMEM  299520

#define JSTEP_BYTES (64 * BATCH * 2048)   // +64 rows of [M][1024] bf16 = 1 MiB

__global__ __launch_bounds__(256) void attn_core(
    const float* __restrict__ qhead,          // [M][1024] f32
    const unsigned short* __restrict__ kh,    // [M][1024] bf16
    const unsigned short* __restrict__ vh,    // [M][1024] bf16
    const unsigned short* __restrict__ kr,    // [M_R][1024] bf16
    const unsigned char* __restrict__ code,   // [B][Q][K]
    const float* __restrict__ rwb, const float* __restrict__ rrb,
    const float* __restrict__ rsb, const float* __restrict__ segemb,
    unsigned short* __restrict__ av,          // [M][1024] bf16 out
    int sbit) {
    extern __shared__ char smem[];
    float*          qf32 = (float*)(smem + OFF_QF32);            // stride 68
    unsigned short* qwT  = (unsigned short*)(smem + OFF_QWT);    // stride 72
    unsigned short* qrT  = (unsigned short*)(smem + OFF_QRT);    // stride 72
    float*          efv  = (float*)(smem + OFF_EF);
    unsigned short* k_s  = (unsigned short*)(smem + OFF_KS);     // stride 72
    unsigned short* kr_s = (unsigned short*)(smem + OFF_KRS);    // stride 72
    float*          bdf  = (float*)(smem + OFF_BDF);             // stride 68
    float*          sc   = (float*)(smem + OFF_SC);              // stride 516
    unsigned short* pb   = (unsigned short*)(smem + OFF_PB);     // stride 520
    float*          redA = (float*)(smem + OFF_RA);
    float*          redB = (float*)(smem + OFF_RB);

    const int tid  = threadIdx.x;
    const int lane = tid & 31;
    const int wave = tid >> 5;
    const int it   = blockIdx.x & 7;
    const int n    = (blockIdx.x >> 3) & 15;
    const int b    = blockIdx.x >> 7;
    const int i0   = it * 64;
    const int hi8  = ((lane >> 4) & 1) << 3;

    const unsigned ksA  = lds_addr_of(k_s);
    const unsigned krsA = lds_addr_of(kr_s);

    // ---- phase 1: q tile + biases -> transposed bf16 operands ----
#pragma unroll
    for (int k = 0; k < 16; ++k) {
        int idx = tid + k * 256;              // 0..4095
        int i = idx >> 6, d = idx & 63;
        float qv = qhead[(((size_t)((i0 + i) * BATCH + b)) << 10) + n * 64 + d];
        qf32[i * 68 + d] = qv;
        qwT[d * 72 + i] = f2bf(qv + rwb[n * 64 + d]);
        qrT[d * 72 + i] = f2bf(qv + rrb[n * 64 + d]);
    }
    __syncthreads();

    // ---- phase 2: ef[i][s] = (q[i]+r_s_bias) . seg_embed[s][n] ----
    if (tid < 128) {
        int i = tid >> 1, s = tid & 1;
        float acc = 0.f;
        for (int d = 0; d < 64; ++d)
            acc += (qf32[i * 68 + d] + rsb[n * 64 + d]) *
                   segemb[(size_t)(s * NHEAD + n) * 64 + d];
        efv[i * 2 + s] = acc;
    }
    __syncthreads();

    // ---- phase 3: transposed score tiles ----
    const int wj = wave & 3;    // 16 j-rows group   (ac output M dim)
    const int wi = wave >> 2;   // 32 i-cols group   (ac output N dim)

    // hoisted per-thread staging addresses (advance by 1 MiB per j-step)
    const int srow = tid >> 2, sc16 = (tid & 3) * 16;
    const unsigned ksOff = ksA + srow * 144 + sc16;
    const char* kP = (const char*)kh +
        (size_t)(srow * BATCH + b) * 2048 + n * 128 + sc16;
    const int krow0 = tid >> 2, krow1 = (tid + 256) >> 2;
    const unsigned krOff0 = krsA + krow0 * 144 + sc16;
    const unsigned krOff1 = krsA + krow1 * 144 + sc16;
    const int pb0 = QLEN - i0 - 64;   // p_base at j0 = 0
    const char* krP0 = (const char*)kr +
        (size_t)((pb0 + krow0) * BATCH + b) * 2048 + n * 128 + sc16;
    const char* krP1 = (const char*)kr +
        (size_t)((pb0 + krow1) * BATCH + b) * 2048 + n * 128 + sc16;

    for (int j0 = 0; j0 < KLEN; j0 += 64) {
        async_load_b128(ksOff, kP);     kP += JSTEP_BYTES;
        async_load_b128(krOff0, krP0);  krP0 += JSTEP_BYTES;
        async_load_b128(krOff1, krP1);  krP1 += JSTEP_BYTES;
        wait_async0();
        __syncthreads();

        // ac^T: wave computes [16 j][32 i] (two 16x16 tiles)
        v8f acc_ac[2] = {};
#pragma unroll
        for (int kc = 0; kc < 2; ++kc) {
            v16bf a = load_frag_a(k_s + (wj * 16) * 72 + kc * 32, 72, lane);
#pragma unroll
            for (int t = 0; t < 2; ++t) {
                v16bf bb = load_frag_b(qwT + (kc * 32) * 72 + wi * 32 + t * 16, 72, lane);
                acc_ac[t] = wmma_bf16(a, bb, acc_ac[t]);
            }
        }
        // bd^T band: wave computes [16 pp][64 i] (four 16x16 tiles)
        v8f acc_bd[4] = {};
#pragma unroll
        for (int kc = 0; kc < 2; ++kc) {
            v16bf a = load_frag_a(kr_s + (wave * 16) * 72 + kc * 32, 72, lane);
#pragma unroll
            for (int t = 0; t < 4; ++t) {
                v16bf bb = load_frag_b(qrT + (kc * 32) * 72 + t * 16, 72, lane);
                acc_bd[t] = wmma_bf16(a, bb, acc_bd[t]);
            }
        }
#pragma unroll
        for (int t = 0; t < 4; ++t)
#pragma unroll
            for (int r = 0; r < 8; ++r) {
                int pp  = wave * 16 + r + hi8;
                int i_c = t * 16 + (lane & 15);
                bdf[pp * 68 + i_c] = acc_bd[t][r];
            }
        __syncthreads();

        // score[i][j] = (ac + band_bd + ef) * SCALE - 1e30*mask
#pragma unroll
        for (int t = 0; t < 2; ++t)
#pragma unroll
            for (int r = 0; r < 8; ++r) {
                int j_l = wj * 16 + r + hi8;
                int i_l = wi * 32 + t * 16 + (lane & 15);
                int j   = j0 + j_l;
                unsigned char cd = code[(size_t)(b * QLEN + i0 + i_l) * KLEN + j];
                float ef   = ((cd >> 2) & 1) ? efv[i_l * 2 + 1] : efv[i_l * 2];
                float mask = (float)((cd >> sbit) & 1);
                float s = (acc_ac[t][r] + bdf[(64 + j_l - i_l) * 68 + i_l] + ef)
                              * ATT_SCALE - 1e30f * mask;
                sc[i_l * 516 + j] = s;
            }
        __syncthreads();
    }

    // ---- phase 4: softmax over 512 cols, 4 threads per row ----
    {
        int row = tid >> 2, part = tid & 3;
        float mx = -3.402823e38f;
        for (int c = 0; c < 128; ++c)
            mx = fmaxf(mx, sc[row * 516 + part * 128 + c]);
        redA[tid] = mx;
        __syncthreads();
        float rmax = fmaxf(fmaxf(redA[row * 4], redA[row * 4 + 1]),
                           fmaxf(redA[row * 4 + 2], redA[row * 4 + 3]));
        float sum = 0.f;
        for (int c = 0; c < 128; ++c) {
            float e = __expf(sc[row * 516 + part * 128 + c] - rmax);
            sc[row * 516 + part * 128 + c] = e;
            sum += e;
        }
        redB[tid] = sum;
        __syncthreads();
        float inv = 1.f / (redB[row * 4] + redB[row * 4 + 1] +
                           redB[row * 4 + 2] + redB[row * 4 + 3]);
        for (int c = 0; c < 128; ++c)
            pb[row * 520 + part * 128 + c] = f2bf(sc[row * 516 + part * 128 + c] * inv);
        __syncthreads();
    }

    // ---- phase 5: out = prob x V, V ping-ponged between k_s and kr_s ----
    const int wm = wave & 3;    // 16 i-rows group
    const int wp = wave >> 2;   // 32 d-cols group
    v8f acc_o[2] = {};
    const char* vP = (const char*)vh +
        (size_t)(srow * BATCH + b) * 2048 + n * 128 + sc16;

    async_load_b128(ksOff, vP);  vP += JSTEP_BYTES;   // stage j0=0 into k_s
    for (int jt = 0; jt < 8; ++jt) {
        const int cur = jt & 1;
        if (jt != 7) {
            // next buffer: cur==0 -> kr_s, cur==1 -> k_s
            async_load_b128((cur ? ksOff : (krsA + srow * 144 + sc16)), vP);
            vP += JSTEP_BYTES;
            wait_async1();
        } else {
            wait_async0();
        }
        __syncthreads();
        const unsigned short* vs = cur ? kr_s : k_s;
        const int j0 = jt * 64;
#pragma unroll
        for (int kc = 0; kc < 2; ++kc) {
            v16bf a = load_frag_a(pb + (wm * 16) * 520 + j0 + kc * 32, 520, lane);
#pragma unroll
            for (int t = 0; t < 2; ++t) {
                v16bf bb = load_frag_b(vs + (kc * 32) * 72 + wp * 32 + t * 16, 72, lane);
                acc_o[t] = wmma_bf16(a, bb, acc_o[t]);
            }
        }
        __syncthreads();
    }

    // ---- phase 6: write attn_vec (bf16, GEMM-ready layout) ----
#pragma unroll
    for (int t = 0; t < 2; ++t)
#pragma unroll
        for (int r = 0; r < 8; ++r) {
            int i_l = wm * 16 + r + hi8;
            int d_l = wp * 32 + t * 16 + (lane & 15);
            av[(((size_t)((i0 + i_l) * BATCH + b)) << 10) + n * 64 + d_l] =
                f2bf(acc_o[t][r]);
        }
}

// ---------------------------------------------------------------------------
__global__ __launch_bounds__(256) void residual_ln(
    const float* __restrict__ proj, const float* __restrict__ inp,
    const float* __restrict__ gamma, const float* __restrict__ beta,
    float* __restrict__ out) {
    __shared__ float sA[256], sB[256];
    int m = blockIdx.x, t = threadIdx.x;
    float x[4], s = 0.f, s2 = 0.f;
#pragma unroll
    for (int k = 0; k < 4; ++k) {
        int d = t + k * 256;
        float v = proj[((size_t)m << 10) + d] + inp[((size_t)m << 10) + d];
        x[k] = v; s += v; s2 += v * v;
    }
    sA[t] = s; sB[t] = s2;
    __syncthreads();
    for (int off = 128; off > 0; off >>= 1) {
        if (t < off) { sA[t] += sA[t + off]; sB[t] += sB[t + off]; }
        __syncthreads();
    }
    float mean = sA[0] * (1.f / DMODEL);
    float var  = sB[0] * (1.f / DMODEL) - mean * mean;
    float inv  = rsqrtf(var + 1e-12f);
#pragma unroll
    for (int k = 0; k < 4; ++k) {
        int d = t + k * 256;
        out[((size_t)m << 10) + d] = (x[k] - mean) * inv * gamma[d] + beta[d];
    }
}

// ---------------------------------------------------------------------------
extern "C" void kernel_launch(void* const* d_in, const int* in_sizes, int n_in,
                              void* d_out, int out_size, void* d_ws, size_t ws_size,
                              hipStream_t stream) {
    (void)in_sizes; (void)n_in; (void)out_size; (void)ws_size;
    const float* h    = (const float*)d_in[0];
    const float* g    = (const float*)d_in[1];
    const float* mh   = (const float*)d_in[2];
    const float* mg   = (const float*)d_in[3];
    const float* r    = (const float*)d_in[4];
    const float* segm = (const float*)d_in[5];
    const float* Wq   = (const float*)d_in[6];
    const float* Wk   = (const float*)d_in[7];
    const float* Wv   = (const float*)d_in[8];
    const float* Wo   = (const float*)d_in[9];
    const float* Wr   = (const float*)d_in[10];
    const float* rwb  = (const float*)d_in[11];
    const float* rrb  = (const float*)d_in[12];
    const float* rsb  = (const float*)d_in[13];
    const float* seme = (const float*)d_in[14];
    const float* lng  = (const float*)d_in[15];
    const float* lnb  = (const float*)d_in[16];
    float* out = (float*)d_out;

    char* w = (char*)d_ws;
    size_t off = 0;
    auto alloc = [&](size_t bytes) -> void* {
        void* p = w + off;
        off += (bytes + 255) & ~(size_t)255;
        return p;
    };
    unsigned short* hb   = (unsigned short*)alloc((size_t)MROWS * DMODEL * 2);
    unsigned short* gb   = (unsigned short*)alloc((size_t)MROWS * DMODEL * 2);
    unsigned short* rb   = (unsigned short*)alloc((size_t)MROWS_R * DMODEL * 2);
    unsigned short* Wqb  = (unsigned short*)alloc((size_t)DMODEL * DMODEL * 2);
    unsigned short* Wkb  = (unsigned short*)alloc((size_t)DMODEL * DMODEL * 2);
    unsigned short* Wvb  = (unsigned short*)alloc((size_t)DMODEL * DMODEL * 2);
    unsigned short* Wrb  = (unsigned short*)alloc((size_t)DMODEL * DMODEL * 2);
    unsigned short* WoT  = (unsigned short*)alloc((size_t)DMODEL * DMODEL * 2);
    unsigned char*  code = (unsigned char*)alloc((size_t)BATCH * QLEN * KLEN);
    float* qh = (float*)alloc((size_t)MROWS * DMODEL * 4);
    float* qg = (float*)alloc((size_t)MROWS * DMODEL * 4);
    unsigned short* khb = (unsigned short*)alloc((size_t)MROWS * DMODEL * 2);
    unsigned short* vhb = (unsigned short*)alloc((size_t)MROWS * DMODEL * 2);
    unsigned short* krb = (unsigned short*)alloc((size_t)MROWS_R * DMODEL * 2);
    unsigned short* avh = (unsigned short*)alloc((size_t)MROWS * DMODEL * 2);
    unsigned short* avg = (unsigned short*)alloc((size_t)MROWS * DMODEL * 2);
    float* poh = (float*)alloc((size_t)MROWS * DMODEL * 4);
    float* pog = (float*)alloc((size_t)MROWS * DMODEL * 4);

    const int nAct = MROWS * DMODEL;     // 4 Mi elems
    const int nR   = MROWS_R * DMODEL;   // 8 Mi
    const int nW   = DMODEL * DMODEL;    // 1 Mi

    cvt_f32_bf16<<<nAct / 1024, 256, 0, stream>>>(h, hb, nAct);
    cvt_f32_bf16<<<nAct / 1024, 256, 0, stream>>>(g, gb, nAct);
    cvt_f32_bf16<<<nR / 1024, 256, 0, stream>>>(r, rb, nR);
    cvt_f32_bf16<<<nW / 1024, 256, 0, stream>>>(Wq, Wqb, nW);
    cvt_f32_bf16<<<nW / 1024, 256, 0, stream>>>(Wk, Wkb, nW);
    cvt_f32_bf16<<<nW / 1024, 256, 0, stream>>>(Wv, Wvb, nW);
    cvt_f32_bf16<<<nW / 1024, 256, 0, stream>>>(Wr, Wrb, nW);
    cvt_transpose_bf16<<<1024, 256, 0, stream>>>(Wo, WoT);   // WoT[p][h]

    pack_code<<<(QLEN * KLEN * BATCH) / 256, 256, 0, stream>>>(mh, mg, segm, code);

    dim3 gdim(DMODEL / 128, MROWS / 128);
    dim3 gdimR(DMODEL / 128, MROWS_R / 128);
    gemm_bf16_128x128<<<gdim, 256, 0, stream>>>(hb, Wqb, qh, nullptr);
    gemm_bf16_128x128<<<gdim, 256, 0, stream>>>(gb, Wqb, qg, nullptr);
    gemm_bf16_128x128<<<gdim, 256, 0, stream>>>(hb, Wkb, nullptr, khb);
    gemm_bf16_128x128<<<gdim, 256, 0, stream>>>(hb, Wvb, nullptr, vhb);
    gemm_bf16_128x128<<<gdimR, 256, 0, stream>>>(rb, Wrb, nullptr, krb);

    const int attBlocks = BATCH * NHEAD * (QLEN / 64);   // 1024
    attn_core<<<attBlocks, 256, ATT_SMEM, stream>>>(
        qh, khb, vhb, krb, code, rwb, rrb, rsb, seme, avh, 0);
    attn_core<<<attBlocks, 256, ATT_SMEM, stream>>>(
        qg, khb, vhb, krb, code, rwb, rrb, rsb, seme, avg, 1);

    gemm_bf16_128x128<<<gdim, 256, 0, stream>>>(avh, WoT, poh, nullptr);
    gemm_bf16_128x128<<<gdim, 256, 0, stream>>>(avg, WoT, pog, nullptr);

    residual_ln<<<MROWS, 256, 0, stream>>>(poh, h, lng, lnb, out);
    residual_ln<<<MROWS, 256, 0, stream>>>(pog, g, lng, lnb, out + (size_t)MROWS * DMODEL);
}
